// VectorQuantizer_12876311953424
// MI455X (gfx1250) — matrix operand order repaired and checked by reading, not compiled, not run on hardware
//
#include <hip/hip_runtime.h>
#include <hip/hip_bf16.h>

#define NUM_EMB 4096
#define EMB_DIM 512
#define N_TOK   32768            // 32 * 1024
#define ROWS_PER_BLK 128         // 8 waves x 16 rows
#define ES_STRIDE 520            // 512 + 8 bf16 pad -> breaks 64-bank conflict
#define ES_ELEMS  (16 * ES_STRIDE)
#define NCHUNK    (NUM_EMB / 16) // 256

typedef __attribute__((ext_vector_type(8)))  __bf16 v8bf;
typedef __attribute__((ext_vector_type(16))) __bf16 v16bf;
typedef __attribute__((ext_vector_type(8)))  float  v8f;

__device__ __forceinline__ unsigned short f2bf(float f) {
    unsigned int u = __float_as_uint(f);
    u += 0x7FFFu + ((u >> 16) & 1u);   // round-to-nearest-even
    return (unsigned short)(u >> 16);
}
__device__ __forceinline__ unsigned int pack2(float a, float b) {
    return (unsigned int)f2bf(a) | ((unsigned int)f2bf(b) << 16);
}

// B fragment: 16 contiguous bf16 from LDS (two 16B ds loads)
__device__ __forceinline__ v16bf loadB(const unsigned short* p) {
    v8bf lo = *(const v8bf*)(p);
    v8bf hi = *(const v8bf*)(p + 8);
    return __builtin_shufflevector(lo, hi,
        0, 1, 2, 3, 4, 5, 6, 7, 8, 9, 10, 11, 12, 13, 14, 15);
}

// Async DMA: 16 bytes global -> LDS, tracked by ASYNCcnt (no VGPR round-trip)
__device__ __forceinline__ void async_copy16(unsigned lds_addr, unsigned goff,
                                             const unsigned short* sbase) {
    asm volatile("global_load_async_to_lds_b128 %0, %1, %2"
                 :: "v"(lds_addr), "v"(goff), "s"(sbase) : "memory");
}
#define ASYNC_WAIT(n) asm volatile("s_wait_asynccnt " #n ::: "memory")

// ---------------- E prep: f32 -> bf16 (row-major, ws) + ||e||^2 ------------
__global__ void vq_prep_kernel(const float* __restrict__ emb,
                               unsigned short* __restrict__ ebf,
                               float* __restrict__ e_norm) {
    __shared__ float red[128];
    const int row = blockIdx.x;
    const int t   = threadIdx.x;                 // 128 threads, 1 float4 each
    float4 v = ((const float4*)(emb + (size_t)row * EMB_DIM))[t];
    unsigned int* dst = (unsigned int*)(ebf + (size_t)row * EMB_DIM + t * 4);
    dst[0] = pack2(v.x, v.y);
    dst[1] = pack2(v.z, v.w);
    red[t] = v.x * v.x + v.y * v.y + v.z * v.z + v.w * v.w;
    __syncthreads();
    for (int o = 64; o > 0; o >>= 1) {
        if (t < o) red[t] += red[t + o];
        __syncthreads();
    }
    if (t == 0) e_norm[row] = red[0];
}

// ---------------- fused bf16-WMMA GEMM + running argmin --------------------
__global__ void __launch_bounds__(256)
vq_argmin_kernel(const float* __restrict__ x,
                 const unsigned short* __restrict__ ebf,
                 const float* __restrict__ e_norm,
                 float* __restrict__ out_idx) {
    extern __shared__ unsigned short smem[];     // 2 x (16 x 520) bf16

    const int t    = threadIdx.x;
    const int row0 = blockIdx.x * ROWS_PER_BLK;
    const int lane = t & 31;
    const int w    = t >> 5;        // wave id: owns rows w*16 .. w*16+15
    const int nloc = lane & 15;
    const int h    = lane >> 4;

    // ---- A operand resident in registers: 16 fragments (128 VGPRs/lane).
    // A k order used here (same in B): elems 0-15 = K 16h..16h+15
    v16bf areg[16];
    {
        const float* xr = x + (size_t)(row0 + w * 16 + nloc) * EMB_DIM;
#pragma unroll
        for (int i = 0; i < 16; ++i) {
            const float* plo = xr + i * 32 + h * 16;
            const float* phi = plo + 8;
            float4 q0 = *(const float4*)(plo);
            float4 q1 = *(const float4*)(plo + 4);
            float4 q2 = *(const float4*)(phi);
            float4 q3 = *(const float4*)(phi + 4);
            unsigned short tmp[16];
            tmp[0]  = f2bf(q0.x); tmp[1]  = f2bf(q0.y);
            tmp[2]  = f2bf(q0.z); tmp[3]  = f2bf(q0.w);
            tmp[4]  = f2bf(q1.x); tmp[5]  = f2bf(q1.y);
            tmp[6]  = f2bf(q1.z); tmp[7]  = f2bf(q1.w);
            tmp[8]  = f2bf(q2.x); tmp[9]  = f2bf(q2.y);
            tmp[10] = f2bf(q2.z); tmp[11] = f2bf(q2.w);
            tmp[12] = f2bf(q3.x); tmp[13] = f2bf(q3.y);
            tmp[14] = f2bf(q3.z); tmp[15] = f2bf(q3.w);
            __builtin_memcpy(&areg[i], tmp, 32);
        }
    }

    const unsigned esb0 = (unsigned)(uintptr_t)(smem);
    const unsigned esb1 = (unsigned)(uintptr_t)(smem + ES_ELEMS);

    // Issue one 16-row bf16 E chunk (16 KB) as 4 async 16B DMAs per thread.
    auto issue_chunk = [&](int kc, unsigned es_base) {
#pragma unroll
        for (int j = 0; j < 4; ++j) {
            int u   = t + j * 256;        // 0..1023 16B units
            int row = u >> 6;             // 0..15
            int c8  = u & 63;             // 16B unit within row
            unsigned lds  = es_base + (unsigned)(row * (ES_STRIDE * 2) + c8 * 16);
            unsigned goff = (unsigned)(((kc * 16 + row) * EMB_DIM + c8 * 8) * 2);
            async_copy16(lds, goff, ebf);
        }
    };

    float minv[8];
    int   mini[8];
#pragma unroll
    for (int r = 0; r < 8; ++r) { minv[r] = 3.4e38f; mini[r] = 0; }

    issue_chunk(0, esb0);                        // prologue: chunk 0 -> buf0

    for (int kc = 0; kc < NCHUNK; ++kc) {
        if (kc + 1 < NCHUNK) {                   // overlap next copy w/ compute
            issue_chunk(kc + 1, (kc + 1) & 1 ? esb1 : esb0);
            ASYNC_WAIT(4);                       // own 4 DMAs of chunk kc done
        } else {
            ASYNC_WAIT(0);
        }
        __syncthreads();                         // all waves' chunk-kc DMAs done

        const unsigned short* erow =
            smem + (kc & 1) * ES_ELEMS + nloc * ES_STRIDE + h * 16;

        v8f acc = {};
        v16bf bcur = loadB(erow);
#pragma unroll
        for (int i = 0; i < 16; ++i) {           // 2 ds_loads + 1 wmma, pipelined
            v16bf bnext;
            if (i < 15) bnext = loadB(erow + (i + 1) * 32);
            acc = __builtin_amdgcn_wmma_f32_16x16x32_bf16(
                false, areg[i], false, bcur, (short)0, acc, false, false);
            if (i < 15) bcur = bnext;
        }

        // score = ||e||^2 - 2 x.e   (||x||^2 constant per row -> dropped)
        const int   n_g = kc * 16 + nloc;
        const float en  = e_norm[n_g];
#pragma unroll
        for (int r = 0; r < 8; ++r) {
            float score = __builtin_fmaf(-2.0f, acc[r], en);
            if (score < minv[r]) { minv[r] = score; mini[r] = n_g; }
        }
        __syncthreads();                         // buf (kc&1) reusable next iter
    }

    // Cross-lane argmin over each 16-lane half (ties -> lowest index)
#pragma unroll
    for (int r = 0; r < 8; ++r) {
        float v = minv[r];
        int   i = mini[r];
#pragma unroll
        for (int s = 1; s < 16; s <<= 1) {
            float ov = __shfl_xor(v, s, 32);
            int   oi = __shfl_xor(i, s, 32);
            if (ov < v || (ov == v && oi < i)) { v = ov; i = oi; }
        }
        if (nloc == 0) {
            int row_g = row0 + w * 16 + h * 8 + r;   // C layout: M = r + 8h
            out_idx[row_g] = (float)i;
        }
    }
}

// ------------- gather quantized + per-row squared-error partial ------------
__global__ void vq_gather_kernel(const float* __restrict__ x,
                                 const float* __restrict__ emb,
                                 const float* __restrict__ idxf,
                                 float* __restrict__ out_q,
                                 float* __restrict__ partial) {
    __shared__ float red[128];
    const int n = blockIdx.x;
    const int t = threadIdx.x;
    const int idx = (int)idxf[n];
    const float4* er = (const float4*)(emb + (size_t)idx * EMB_DIM);
    const float4* xr = (const float4*)(x + (size_t)n * EMB_DIM);
    float4*       qr = (float4*)(out_q + (size_t)n * EMB_DIM);
    float4 e  = er[t];
    float4 xv = xr[t];
    qr[t] = e;
    float dx = e.x - xv.x, dy = e.y - xv.y, dz = e.z - xv.z, dw = e.w - xv.w;
    red[t] = dx * dx + dy * dy + dz * dz + dw * dw;
    __syncthreads();
    for (int o = 64; o > 0; o >>= 1) {
        if (t < o) red[t] += red[t + o];
        __syncthreads();
    }
    if (t == 0) partial[n] = red[0];
}

// ---------------- deterministic final loss reduction -----------------------
__global__ void vq_loss_kernel(const float* __restrict__ partial,
                               float* __restrict__ out_loss) {
    __shared__ float red[256];
    const int t = threadIdx.x;
    float s = 0.f;
    for (int i = t; i < N_TOK; i += 256) s += partial[i];
    red[t] = s;
    __syncthreads();
    for (int o = 128; o > 0; o >>= 1) {
        if (t < o) red[t] += red[t + o];
        __syncthreads();
    }
    // loss = q_latent + 0.25 * e_latent = 1.25 * MSE (forward values identical)
    if (t == 0) out_loss[0] = red[0] * (1.25f / ((float)N_TOK * (float)EMB_DIM));
}

extern "C" void kernel_launch(void* const* d_in, const int* in_sizes, int n_in,
                              void* d_out, int out_size, void* d_ws, size_t ws_size,
                              hipStream_t stream) {
    const float* x   = (const float*)d_in[0];   // [32,1024,512] f32
    const float* emb = (const float*)d_in[1];   // [4096,512] f32

    float* out      = (float*)d_out;
    float* out_loss = out;                                   // [1]
    float* out_q    = out + 1;                               // [N_TOK*EMB_DIM]
    float* out_idx  = out + 1 + (size_t)N_TOK * EMB_DIM;     // [N_TOK] (as float)

    // workspace: bf16 codebook (4 MB, L2-resident) | e_norm | partials
    unsigned short* ebf     = (unsigned short*)d_ws;
    float*          e_norm  = (float*)(ebf + (size_t)NUM_EMB * EMB_DIM);
    float*          partial = e_norm + NUM_EMB;

    vq_prep_kernel<<<NUM_EMB, 128, 0, stream>>>(emb, ebf, e_norm);

    size_t lds = (size_t)(2 * ES_ELEMS) * sizeof(unsigned short);  // 33.3 KB
    vq_argmin_kernel<<<N_TOK / ROWS_PER_BLK, 256, lds, stream>>>(
        x, ebf, e_norm, out_idx);

    vq_gather_kernel<<<N_TOK, 128, 0, stream>>>(x, emb, out_idx, out_q, partial);
    vq_loss_kernel<<<1, 256, 0, stream>>>(partial, out_loss);
}
